// MultiHeadAttention_76055280878050
// MI455X (gfx1250) — compile-verified, matrix operand order checked by reading, not run on hardware
//
#include <hip/hip_runtime.h>
#include <hip/hip_fp16.h>

#ifndef __has_builtin
#define __has_builtin(x) 0
#endif
#if __has_builtin(__builtin_amdgcn_global_load_async_to_lds_b128)
#define USE_ASYNC_LDS 1
#else
#define USE_ASYNC_LDS 0
#endif

#define AS1 __attribute__((address_space(1)))
#define AS3 __attribute__((address_space(3)))

typedef __attribute__((ext_vector_type(16))) _Float16 v16h;
typedef __attribute__((ext_vector_type(8)))  _Float16 v8h;
typedef __attribute__((ext_vector_type(4)))  _Float16 v4h;
typedef __attribute__((ext_vector_type(8)))  float    v8f;
typedef __attribute__((ext_vector_type(4)))  int      v4i;

typedef AS1 v4i* async_gptr;   // global source for async-to-LDS (per diagnostic)
typedef AS3 v4i* async_lptr;   // LDS destination

#define SEQ    2048
#define EMBED  1024
#define HEADS  16
#define HDIM   64
#define BATCH  2
#define ATT_SCALE 0.125f   // 64^-0.5

union Frag { v16h f; v8h h[2]; };

static __device__ inline v8f wmma_f16(v16h a, v16h b, v8f c) {
  // D(f32 16x16) = A(f16 16x32) * B(f16 32x16) + C
  return __builtin_amdgcn_wmma_f32_16x16x32_f16(false, a, false, b, (short)0, c,
                                                false, false);
}

static __device__ inline v8f vzero8() {
  v8f z;
#pragma unroll
  for (int i = 0; i < 8; ++i) z[i] = 0.0f;
  return z;
}

// ---------------------------------------------------------------------------
// Kernel 1: QKV projection.  X[4096,1024]f32 @ Wqkv[1024,3072]f32 + b ->
// Q,K f16 head-major [B,H,N,Dh] (Q pre-scaled by 1/sqrt(Dh));
// V f16 head-major TRANSPOSED [B,H,Dh,N] so the attention kernel can copy it
// to LDS with b128 moves (async-to-LDS friendly).
// Block = 256 thr (8 waves), tile 128x128, K-step 32, double-buffered LDS.
// ---------------------------------------------------------------------------
__global__ __launch_bounds__(256) void qkv_gemm_kernel(
    const float* __restrict__ X, const float* __restrict__ W,
    const float* __restrict__ bias,
    _Float16* __restrict__ Qo, _Float16* __restrict__ Ko,
    _Float16* __restrict__ Vo)
{
  __shared__ _Float16 ldsA[2][128][40];   // [m][k], 16B-aligned rows
  __shared__ _Float16 ldsBt[2][128][40];  // [n][k] (N-major)

  const int tid  = threadIdx.x;
  const int lane = tid & 31, wid = tid >> 5;
  const int wm = wid & 1, wn = wid >> 1;
  const int ln = lane & 15, hi = lane >> 4;
  const int m0 = blockIdx.y * 128;
  const int n0 = blockIdx.x * 128;
  const int kb4 = tid >> 5;   // B tile: 4-row k block
  const int nb4 = tid & 31;   // B tile: 4-col n block

  v8f acc[4][2];
#pragma unroll
  for (int i = 0; i < 4; ++i)
#pragma unroll
    for (int j = 0; j < 2; ++j) acc[i][j] = vzero8();

  float pa[4][4], pb[4][4];

  auto load_tiles = [&](int k0) {
#pragma unroll
    for (int i = 0; i < 4; ++i) {     // A: 128x32 f32, 4 float4 per thread
      const int f = tid + i * 256;
      const float4 v =
          *(const float4*)(X + (size_t)(m0 + (f >> 3)) * EMBED + k0 + (f & 7) * 4);
      pa[i][0] = v.x; pa[i][1] = v.y; pa[i][2] = v.z; pa[i][3] = v.w;
    }
#pragma unroll
    for (int i = 0; i < 4; ++i) {     // B: 4x4 f32 block per thread
      const float4 v =
          *(const float4*)(W + (size_t)(k0 + kb4 * 4 + i) * 3072 + n0 + nb4 * 4);
      pb[i][0] = v.x; pb[i][1] = v.y; pb[i][2] = v.z; pb[i][3] = v.w;
    }
  };
  auto store_tiles = [&](int buf) {
#pragma unroll
    for (int i = 0; i < 4; ++i) {
      const int f = tid + i * 256;
      v4h h; h.x = (_Float16)pa[i][0]; h.y = (_Float16)pa[i][1];
      h.z = (_Float16)pa[i][2]; h.w = (_Float16)pa[i][3];
      *(v4h*)&ldsA[buf][f >> 3][(f & 7) * 4] = h;
    }
#pragma unroll
    for (int jj = 0; jj < 4; ++jj) {  // register transpose -> b64 stores
      v4h h; h.x = (_Float16)pb[0][jj]; h.y = (_Float16)pb[1][jj];
      h.z = (_Float16)pb[2][jj]; h.w = (_Float16)pb[3][jj];
      *(v4h*)&ldsBt[buf][nb4 * 4 + jj][kb4 * 4] = h;
    }
  };

  load_tiles(0);
  store_tiles(0);
  __syncthreads();
  int buf = 0;

  for (int k0 = 0; k0 < EMBED; k0 += 32) {
    const bool has_next = (k0 + 32) < EMBED;
    if (has_next) {
      load_tiles(k0 + 32);
      if ((k0 + 64) < EMBED) {
        __builtin_prefetch(X + (size_t)(m0 + (tid >> 3)) * EMBED + k0 + 64, 0, 1);
        __builtin_prefetch(W + (size_t)(k0 + 64 + kb4 * 4) * 3072 + n0 + nb4 * 4, 0, 1);
      }
    }
    Frag a[4], b[2];
#pragma unroll
    for (int i = 0; i < 4; ++i) {
      const int row = wm * 64 + i * 16 + ln;
      a[i].h[0] = *(const v8h*)&ldsA[buf][row][hi * 8];
      a[i].h[1] = *(const v8h*)&ldsA[buf][row][16 + hi * 8];
    }
#pragma unroll
    for (int j = 0; j < 2; ++j) {
      const int row = wn * 32 + j * 16 + ln;
      b[j].h[0] = *(const v8h*)&ldsBt[buf][row][hi * 8];
      b[j].h[1] = *(const v8h*)&ldsBt[buf][row][16 + hi * 8];
    }
#pragma unroll
    for (int i = 0; i < 4; ++i)
#pragma unroll
      for (int j = 0; j < 2; ++j)
        acc[i][j] = wmma_f16(a[i].f, b[j].f, acc[i][j]);
    if (has_next) store_tiles(buf ^ 1);
    __syncthreads();
    buf ^= 1;
  }

  // Epilogue: bias, split Q/K/V, Q scaled, V transposed.
#pragma unroll
  for (int i = 0; i < 4; ++i)
#pragma unroll
    for (int j = 0; j < 2; ++j)
#pragma unroll
      for (int r = 0; r < 8; ++r) {
        const int mg = m0 + wm * 64 + i * 16 + r + 8 * hi;  // b*SEQ + nseq
        const int ng = n0 + wn * 32 + j * 16 + ln;          // 0..3071
        const float val = acc[i][j][r] + bias[ng];
        const int b_ = mg >> 11, ns = mg & 2047;
        const int sec = ng >> 10, e = ng & 1023;
        const int h = e >> 6, d = e & 63;
        const size_t hb = (size_t)(b_ * HEADS + h);
        if (sec == 0)
          Qo[(hb * SEQ + ns) * HDIM + d] = (_Float16)(val * ATT_SCALE);
        else if (sec == 1)
          Ko[(hb * SEQ + ns) * HDIM + d] = (_Float16)val;
        else
          Vo[(hb * HDIM + d) * SEQ + ns] = (_Float16)val;  // transposed
      }
}

// ---------------------------------------------------------------------------
// Kernel 2: flash attention.  One block = (batch*head, 64-query tile).
// 128 thr (4 waves); 16 query rows per wave. K/V tiles double-buffered in LDS
// via async-to-LDS (ASYNCcnt) when available; online softmax; O = P*V.
// ---------------------------------------------------------------------------
__global__ __launch_bounds__(128) void flash_attn_kernel(
    const _Float16* __restrict__ Q, const _Float16* __restrict__ Km,
    const _Float16* __restrict__ Vt, _Float16* __restrict__ O)
{
  __shared__ _Float16 ldsK[2][64][72];    // [key][d]  == N-major B for Q*K^T
  __shared__ _Float16 ldsV[2][64][72];    // [d][key]  == N-major B for P*V
  __shared__ _Float16 ldsP[4][16][72];    // per-wave P tile [m][key]

  const int tid = threadIdx.x;
  const int lane = tid & 31, wid = tid >> 5;
  const int ln = lane & 15, hi = lane >> 4;
  const int bh = blockIdx.x;
  const int q0 = blockIdx.y * 64;
  const size_t kqbase = (size_t)bh * SEQ * HDIM;  // Q,K: [n][d]
  const size_t vbase  = (size_t)bh * HDIM * SEQ;  // Vt:  [d][n]

  // Persistent Q A-fragments (Dh=64 -> 2 k-steps)
  Frag aq[2];
  {
    const _Float16* qrow = Q + kqbase + (size_t)(q0 + wid * 16 + ln) * HDIM;
#pragma unroll
    for (int ks = 0; ks < 2; ++ks) {
      aq[ks].h[0] = *(const v8h*)(qrow + ks * 32 + hi * 8);
      aq[ks].h[1] = *(const v8h*)(qrow + ks * 32 + 16 + hi * 8);
    }
  }

  float mrun[8], lrun[8];
  v8f o4[4];
#pragma unroll
  for (int r = 0; r < 8; ++r) { mrun[r] = -1e30f; lrun[r] = 0.0f; }
#pragma unroll
  for (int j = 0; j < 4; ++j) o4[j] = vzero8();

  auto issue_tile = [&](int kb, int bufi) {
#pragma unroll
    for (int i = 0; i < 4; ++i) {
      const int f = tid + i * 128;
      const int r = f >> 3, c8 = f & 7;
      const _Float16* gk = Km + kqbase + (size_t)(kb + r) * HDIM + c8 * 8;
      const _Float16* gv = Vt + vbase + (size_t)r * SEQ + kb + c8 * 8;
#if USE_ASYNC_LDS
      __builtin_amdgcn_global_load_async_to_lds_b128(
          (async_gptr)gk, (async_lptr)&ldsK[bufi][r][c8 * 8], 0, 0);
      __builtin_amdgcn_global_load_async_to_lds_b128(
          (async_gptr)gv, (async_lptr)&ldsV[bufi][r][c8 * 8], 0, 0);
#else
      *(v8h*)&ldsK[bufi][r][c8 * 8] = *(const v8h*)gk;
      *(v8h*)&ldsV[bufi][r][c8 * 8] = *(const v8h*)gv;
#endif
    }
  };

  issue_tile(0, 0);
  int buf = 0;

  for (int kb = 0; kb < SEQ; kb += 64) {
    const bool has_next = (kb + 64) < SEQ;
    if (has_next) issue_tile(kb + 64, buf ^ 1);   // prefetch next tile
#if USE_ASYNC_LDS
    if (has_next) asm volatile("s_wait_asynccnt 0x8" ::: "memory");
    else          asm volatile("s_wait_asynccnt 0x0" ::: "memory");
#endif
    __syncthreads();

    // S = Q * K^T  (4 key subtiles x 2 k-steps)
    v8f s[4];
#pragma unroll
    for (int j = 0; j < 4; ++j) {
      s[j] = vzero8();
      const int row = j * 16 + ln;
#pragma unroll
      for (int ks = 0; ks < 2; ++ks) {
        Frag bk;
        bk.h[0] = *(const v8h*)&ldsK[buf][row][ks * 32 + hi * 8];
        bk.h[1] = *(const v8h*)&ldsK[buf][row][ks * 32 + 16 + hi * 8];
        s[j] = wmma_f16(aq[ks].f, bk.f, s[j]);
      }
    }

    // Online softmax (rows live on VGPR index r within each 16-lane half)
    float mnew[8], sc[8], psum[8];
#pragma unroll
    for (int r = 0; r < 8; ++r) {
      float mc = fmaxf(fmaxf(s[0][r], s[1][r]), fmaxf(s[2][r], s[3][r]));
#pragma unroll
      for (int off = 1; off <= 8; off <<= 1)
        mc = fmaxf(mc, __shfl_xor(mc, off, 32));
      mnew[r] = fmaxf(mrun[r], mc);
      sc[r] = __expf(mrun[r] - mnew[r]);
      mrun[r] = mnew[r];
      psum[r] = 0.0f;
    }
#pragma unroll
    for (int j = 0; j < 4; ++j)
#pragma unroll
      for (int r = 0; r < 8; ++r) {
        const float p = __expf(s[j][r] - mnew[r]);
        psum[r] += p;
        ldsP[wid][r + 8 * hi][j * 16 + ln] = (_Float16)p;
      }
#pragma unroll
    for (int r = 0; r < 8; ++r) {
      float ps = psum[r];
#pragma unroll
      for (int off = 1; off <= 8; off <<= 1) ps += __shfl_xor(ps, off, 32);
      lrun[r] = lrun[r] * sc[r] + ps;
#pragma unroll
      for (int j = 0; j < 4; ++j) o4[j][r] *= sc[r];
    }
    asm volatile("s_wait_dscnt 0" ::: "memory");  // own-wave P visibility

    // O += P * V
    Frag ap[2];
#pragma unroll
    for (int ks = 0; ks < 2; ++ks) {
      ap[ks].h[0] = *(const v8h*)&ldsP[wid][ln][ks * 32 + hi * 8];
      ap[ks].h[1] = *(const v8h*)&ldsP[wid][ln][ks * 32 + 16 + hi * 8];
    }
#pragma unroll
    for (int j = 0; j < 4; ++j) {
      const int row = j * 16 + ln;
#pragma unroll
      for (int ks = 0; ks < 2; ++ks) {
        Frag bv;
        bv.h[0] = *(const v8h*)&ldsV[buf][row][ks * 32 + hi * 8];
        bv.h[1] = *(const v8h*)&ldsV[buf][row][ks * 32 + 16 + hi * 8];
        o4[j] = wmma_f16(ap[ks].f, bv.f, o4[j]);
      }
    }
    __syncthreads();
    buf ^= 1;
  }

  // Normalize, write token-major f16 output
  const int b_ = bh >> 4, h = bh & 15;
#pragma unroll
  for (int j = 0; j < 4; ++j)
#pragma unroll
    for (int r = 0; r < 8; ++r) {
      const int m = r + 8 * hi;
      const int qidx = q0 + wid * 16 + m;
      const float val = o4[j][r] / lrun[r];
      O[((size_t)(b_ * SEQ + qidx)) * EMBED + h * HDIM + j * 16 + ln] =
          (_Float16)val;
    }
}

// ---------------------------------------------------------------------------
// Kernel 3: output projection.  A[4096,1024]f16 @ Wout[1024,1024]f32 + b -> f32
// Same double-buffered tiling as kernel 1 (A already f16).
// ---------------------------------------------------------------------------
__global__ __launch_bounds__(256) void out_gemm_kernel(
    const _Float16* __restrict__ Ain, const float* __restrict__ W,
    const float* __restrict__ bias, float* __restrict__ out)
{
  __shared__ _Float16 ldsA[2][128][40];
  __shared__ _Float16 ldsBt[2][128][40];

  const int tid  = threadIdx.x;
  const int lane = tid & 31, wid = tid >> 5;
  const int wm = wid & 1, wn = wid >> 1;
  const int ln = lane & 15, hi = lane >> 4;
  const int m0 = blockIdx.y * 128;
  const int n0 = blockIdx.x * 128;
  const int kb4 = tid >> 5;
  const int nb4 = tid & 31;

  v8f acc[4][2];
#pragma unroll
  for (int i = 0; i < 4; ++i)
#pragma unroll
    for (int j = 0; j < 2; ++j) acc[i][j] = vzero8();

  v8h pa[2];
  float pb[4][4];

  auto load_tiles = [&](int k0) {
#pragma unroll
    for (int i = 0; i < 2; ++i) {     // A: f16 128x32, 2 v8h per thread
      const int f = tid + i * 256;
      pa[i] = *(const v8h*)(Ain + (size_t)(m0 + (f >> 2)) * EMBED + k0 + (f & 3) * 8);
    }
#pragma unroll
    for (int i = 0; i < 4; ++i) {     // B: 4x4 f32 block per thread
      const float4 v =
          *(const float4*)(W + (size_t)(k0 + kb4 * 4 + i) * EMBED + n0 + nb4 * 4);
      pb[i][0] = v.x; pb[i][1] = v.y; pb[i][2] = v.z; pb[i][3] = v.w;
    }
  };
  auto store_tiles = [&](int buf) {
#pragma unroll
    for (int i = 0; i < 2; ++i) {
      const int f = tid + i * 256;
      *(v8h*)&ldsA[buf][f >> 2][(f & 3) * 8] = pa[i];
    }
#pragma unroll
    for (int jj = 0; jj < 4; ++jj) {
      v4h h; h.x = (_Float16)pb[0][jj]; h.y = (_Float16)pb[1][jj];
      h.z = (_Float16)pb[2][jj]; h.w = (_Float16)pb[3][jj];
      *(v4h*)&ldsBt[buf][nb4 * 4 + jj][kb4 * 4] = h;
    }
  };

  load_tiles(0);
  store_tiles(0);
  __syncthreads();
  int buf = 0;

  for (int k0 = 0; k0 < EMBED; k0 += 32) {
    const bool has_next = (k0 + 32) < EMBED;
    if (has_next) {
      load_tiles(k0 + 32);
      if ((k0 + 64) < EMBED) {
        __builtin_prefetch(Ain + (size_t)(m0 + (tid >> 2)) * EMBED + k0 + 64, 0, 1);
        __builtin_prefetch(W + (size_t)(k0 + 64 + kb4 * 4) * EMBED + n0 + nb4 * 4, 0, 1);
      }
    }
    Frag a[4], b[2];
#pragma unroll
    for (int i = 0; i < 4; ++i) {
      const int row = wm * 64 + i * 16 + ln;
      a[i].h[0] = *(const v8h*)&ldsA[buf][row][hi * 8];
      a[i].h[1] = *(const v8h*)&ldsA[buf][row][16 + hi * 8];
    }
#pragma unroll
    for (int j = 0; j < 2; ++j) {
      const int row = wn * 32 + j * 16 + ln;
      b[j].h[0] = *(const v8h*)&ldsBt[buf][row][hi * 8];
      b[j].h[1] = *(const v8h*)&ldsBt[buf][row][16 + hi * 8];
    }
#pragma unroll
    for (int i = 0; i < 4; ++i)
#pragma unroll
      for (int j = 0; j < 2; ++j)
        acc[i][j] = wmma_f16(a[i].f, b[j].f, acc[i][j]);
    if (has_next) store_tiles(buf ^ 1);
    __syncthreads();
    buf ^= 1;
  }

#pragma unroll
  for (int i = 0; i < 4; ++i)
#pragma unroll
    for (int j = 0; j < 2; ++j)
#pragma unroll
      for (int r = 0; r < 8; ++r) {
        const int mg = m0 + wm * 64 + i * 16 + r + 8 * hi;
        const int ng = n0 + wn * 32 + j * 16 + ln;
        out[(size_t)mg * EMBED + ng] = acc[i][j][r] + bias[ng];
      }
}

// ---------------------------------------------------------------------------
extern "C" void kernel_launch(void* const* d_in, const int* in_sizes, int n_in,
                              void* d_out, int out_size, void* d_ws, size_t ws_size,
                              hipStream_t stream) {
  (void)in_sizes; (void)n_in; (void)out_size; (void)ws_size;
  const float* x     = (const float*)d_in[0];
  const float* W_qkv = (const float*)d_in[1];
  const float* b_qkv = (const float*)d_in[2];
  const float* W_out = (const float*)d_in[3];
  const float* b_out = (const float*)d_in[4];
  float* out = (float*)d_out;

  const size_t TEN = (size_t)BATCH * SEQ * EMBED;  // 4,194,304 elements
  char* ws = (char*)d_ws;
  _Float16* Qh = (_Float16*)(ws);
  _Float16* Kh = (_Float16*)(ws + TEN * 2);
  _Float16* Vh = (_Float16*)(ws + TEN * 4);  // transposed [B,H,Dh,N]
  _Float16* Ah = (_Float16*)(ws + TEN * 6);

  qkv_gemm_kernel<<<dim3(3072 / 128, (BATCH * SEQ) / 128), 256, 0, stream>>>(
      x, W_qkv, b_qkv, Qh, Kh, Vh);
  flash_attn_kernel<<<dim3(BATCH * HEADS, SEQ / 64), 128, 0, stream>>>(
      Qh, Kh, Vh, Ah);
  out_gemm_kernel<<<dim3(EMBED / 128, (BATCH * SEQ) / 128), 256, 0, stream>>>(
      Ah, W_out, b_out, out);
}
